// EncoderLayer_66967130079316
// MI455X (gfx1250) — matrix operand order repaired and checked by reading, not compiled, not run on hardware
//
#include <hip/hip_runtime.h>

// ---------------------------------------------------------------------------
// Types for CDNA5 WMMA (wave32, 16x16x32 bf16 -> f32)
// ---------------------------------------------------------------------------
typedef __bf16 v16bf __attribute__((ext_vector_type(16)));
typedef __bf16 v8bf  __attribute__((ext_vector_type(8)));
typedef float  v8f   __attribute__((ext_vector_type(8)));
typedef unsigned int v4u  __attribute__((ext_vector_type(4)));
typedef int          v4i  __attribute__((ext_vector_type(4)));
typedef int          v8i  __attribute__((ext_vector_type(8)));
typedef unsigned int v4ui __attribute__((ext_vector_type(4)));

#define DMODEL 1024
#define NHEADS 16
#define DK     64
#define DFF    4096
#define SEQ    2048
#define BATCH  4
#define MROWS  (BATCH * SEQ)   // 8192

// --- CDNA5 TDM feature detection -------------------------------------------
#if defined(__has_builtin)
#if __has_builtin(__builtin_amdgcn_tensor_load_to_lds)
#define HAVE_TDM 1
#endif
#if __has_builtin(__builtin_amdgcn_s_wait_tensorcnt)
#define HAVE_TENSORCNT 1
#endif
#endif
#if defined(HAVE_TDM) && __has_include(<hip/amd_detail/amd_gfx1250_TDM.h>)
#define TDM_6ARG 1
#endif

// ---------------------------------------------------------------------------
// Helpers
// ---------------------------------------------------------------------------
__device__ __forceinline__ unsigned short f2bf(float f) {
  unsigned int u = __float_as_uint(f);
  u += 0x7FFFu + ((u >> 16) & 1u);   // round-to-nearest-even
  return (unsigned short)(u >> 16);
}

// A-fragment (16x32 bf16, MxK): lane = half*16 + (m%16); lane holds
// K = base..base+15 contiguous where base = k0 + half*16.
__device__ __forceinline__ v16bf load_a_frag(const unsigned short* A, int lda,
                                             int m0, int k0, int lane) {
  int row  = m0 + (lane & 15);
  int half = lane >> 4;
  const __bf16* p = (const __bf16*)(A + (size_t)row * lda + k0 + half * 16);
  v8bf lo = *(const v8bf*)p;          // 16B contiguous
  v8bf hi = *(const v8bf*)(p + 8);    // 16B contiguous
  return __builtin_shufflevector(lo, hi, 0,1,2,3,4,5,6,7,8,9,10,11,12,13,14,15);
}

// B-fragment (32x16 bf16, KxN): lane holds column n = n0 + lane%16,
// K = k0 + half*16 .. +15 contiguous (Bp: "row n has contiguous K").
__device__ __forceinline__ v16bf load_b_frag(const unsigned short* Bp, int ldb,
                                             int n0, int k0, int lane) {
  int col  = n0 + (lane & 15);
  int half = lane >> 4;
  return *(const v16bf*)(Bp + (size_t)col * ldb + k0 + half * 16);  // 32B
}

__device__ __forceinline__ v8f wmma_bf16(v16bf a, v16bf b, v8f c) {
  return __builtin_amdgcn_wmma_f32_16x16x32_bf16(false, a, false, b,
                                                 (short)0, c, false, false);
}

__device__ __forceinline__ void wait_tensorcnt0() {
#if defined(HAVE_TENSORCNT)
  __builtin_amdgcn_s_wait_tensorcnt(0);
#elif defined(HAVE_TDM)
  asm volatile("s_wait_tensorcnt 0x0" ::: "memory");
#endif
}

#if defined(HAVE_TDM)
// Issue a TDM 2-D tile load: tile [rows x 32] bf16 at (row m0, col k0) of a
// [Mdim x Kdim] bf16 matrix -> contiguous LDS at ldsOff.
// D# packing per CDNA5 ISA 8.3/8.4: group0 {count,lds_addr,global_addr,type},
// group1 {data_size, tensor_dim0/1, tile_dim0/1, tensor_dim0_stride}.
__device__ __forceinline__ void tdm_load_tile(const unsigned short* gptr,
                                              unsigned ldsOff, int Kdim,
                                              int Mdim, int rows) {
  unsigned long long ga = (unsigned long long)(size_t)gptr;
  v4u g0;
  g0[0] = 1u;                                    // count=1, user mode
  g0[1] = ldsOff;                                // lds_addr (bytes)
  g0[2] = (unsigned)(ga & 0xFFFFFFFFull);        // global_addr[31:0]
  g0[3] = (unsigned)((ga >> 32) & 0x1FFFFFFull)  // global_addr[56:32]
        | (2u << 30);                            // type=2 (image)
  v8i g1;
  unsigned uK = (unsigned)Kdim, uM = (unsigned)Mdim;
  g1[0] = (int)(1u << 16);                       // data_size=2B, wg_mask=0
  g1[1] = (int)((uK & 0xFFFFu) << 16);           // tensor_dim0[15:0] @ bit48
  g1[2] = (int)((uK >> 16) | ((uM & 0xFFFFu) << 16));   // dim0 hi | dim1 lo
  g1[3] = (int)((uM >> 16) | (32u << 16));       // dim1 hi | tile_dim0=32
  g1[4] = (int)((unsigned)rows);                 // tile_dim1=rows, tile_dim2=0
  g1[5] = (int)uK;                               // tensor_dim0_stride[31:0]
  g1[6] = 0;                                     // stride hi | dim1_stride lo
  g1[7] = 0;
  v4i z4 = {0, 0, 0, 0};
#if defined(TDM_6ARG)
  v8i z8 = {0, 0, 0, 0, 0, 0, 0, 0};
  __builtin_amdgcn_tensor_load_to_lds(g0, g1, z4, z4, z8, 0);
#else
  __builtin_amdgcn_tensor_load_to_lds(g0, g1, z4, z4, 0);
#endif
}
#endif  // HAVE_TDM

// ---------------------------------------------------------------------------
// f32 -> bf16 conversion
// ---------------------------------------------------------------------------
__global__ void cvt_f32_bf16(const float* __restrict__ in,
                             unsigned short* __restrict__ out, size_t n) {
  size_t i = (size_t)blockIdx.x * blockDim.x + threadIdx.x;
  size_t st = (size_t)gridDim.x * blockDim.x;
  for (; i < n; i += st) out[i] = f2bf(in[i]);
}

// ---------------------------------------------------------------------------
// GEMM: out = A[M,K](bf16) @ W[N,K](bf16)^T + bias, fused epilogues.
// Block = 8 waves; block tile 128(M) x 128(N); wave tile 32(M) x 64(N).
// A block-tile [128 x 32] staged in LDS via TDM (double-buffered),
// B streamed from global (contiguous 32B fragment loads) + prefetch.
// 8 WMMA per 32-K step per wave.
// MODE 0: bf16 out [M,N]
// MODE 1: bf16 out, V head-transpose layout [B,H,Dk,S]
// MODE 2: f32 out = acc + bias + res (residual add)
// MODE 3: bf16 out, ReLU
// ---------------------------------------------------------------------------
template <int MODE>
__global__ void __launch_bounds__(256)
gemm_bf16_wmma(const unsigned short* __restrict__ A,
               const unsigned short* __restrict__ W,
               const float* __restrict__ bias,
               void* __restrict__ out,
               const float* __restrict__ res,
               int M, int N, int K) {
  const int tid  = threadIdx.x;
  const int lane = tid & 31;
  const int wv   = tid >> 5;
  const int wm   = wv >> 1;          // 0..3  (32 rows each)
  const int wn   = wv & 1;           // 0..1  (64 cols each)
  const int m0blk = blockIdx.x * 128;
  const int mtile = wm * 32;                      // row offset inside LDS tile
  const int ntile = blockIdx.y * 128 + wn * 64;

  __shared__ unsigned short As[2][128 * 32];      // 2 x 8KB double buffer

  v8f acc[2][4];
  #pragma unroll
  for (int g = 0; g < 2; g++)
    #pragma unroll
    for (int j = 0; j < 4; j++) { v8f z = {}; acc[g][j] = z; }

  const int nsteps = K >> 5;

#if defined(HAVE_TDM)
  if (tid == 0)
    tdm_load_tile(A + (size_t)m0blk * K, (unsigned)(size_t)&As[0][0], K, M, 128);
#endif

  for (int s = 0; s < nsteps; ++s) {
    const int k0  = s << 5;
#if defined(HAVE_TDM)
    const int buf = s & 1;
    wait_tensorcnt0();
    __syncthreads();
    if (tid == 0 && s + 1 < nsteps)
      tdm_load_tile(A + (size_t)m0blk * K + (k0 + 32),
                    (unsigned)(size_t)&As[buf ^ 1][0], K, M, 128);
#else
    const int buf = 0;
    // cooperative staging: 128x32 bf16 = 512 x 16B, 2 vectors per thread
    for (int v = tid; v < 512; v += 256) {
      int r = v >> 2;
      int c = (v & 3) * 8;
      *(v4ui*)&As[0][r * 32 + c] =
          *(const v4ui*)&A[(size_t)(m0blk + r) * K + k0 + c];
    }
    __syncthreads();
#endif

    // prefetch next K-slice of the weight rows this lane streams
    if (k0 + 32 < K)
      __builtin_prefetch(W + (size_t)(ntile + (lane & 15)) * K + k0 + 32, 0, 1);

    v16bf a0 = load_a_frag(&As[buf][(size_t)mtile * 32], 32, 0, 0, lane);
    v16bf a1 = load_a_frag(&As[buf][(size_t)(mtile + 16) * 32], 32, 0, 0, lane);
    #pragma unroll
    for (int j = 0; j < 4; j++) {
      v16bf b = load_b_frag(W, K, ntile + j * 16, k0, lane);
      acc[0][j] = wmma_bf16(a0, b, acc[0][j]);
      acc[1][j] = wmma_bf16(a1, b, acc[1][j]);
    }
    __syncthreads();
  }

  const int half = lane >> 4;
  const int nl   = lane & 15;
  #pragma unroll
  for (int g = 0; g < 2; g++) {
    #pragma unroll
    for (int j = 0; j < 4; j++) {
      int n = ntile + j * 16 + nl;
      float bvv = bias[n];
      #pragma unroll
      for (int r = 0; r < 8; r++) {
        int m = m0blk + mtile + g * 16 + half * 8 + r;
        float v = acc[g][j][r] + bvv;
        if (MODE == 3) v = fmaxf(v, 0.0f);
        if (MODE == 2) {
          float* o = (float*)out;
          size_t idx = (size_t)m * N + n;
          o[idx] = v + res[idx];
        } else if (MODE == 1) {
          int hh = n >> 6, d = n & 63;       // Dk = 64
          int b = m >> 11, sq = m & 2047;    // S  = 2048
          unsigned short* o = (unsigned short*)out;
          o[(((size_t)b * NHEADS + hh) * DK + d) * SEQ + sq] = f2bf(v);
        } else {
          unsigned short* o = (unsigned short*)out;
          o[(size_t)m * N + n] = f2bf(v);
        }
      }
    }
  }
}

// ---------------------------------------------------------------------------
// Fused attention: one workgroup per (b, h, 16-query tile).
// Phase 1: scores = QK^T/8 (+mask) -> fp32 strip [16, S] in LDS (WMMA)
// Phase 2: fp32 softmax in LDS, emit P as bf16 [16, S] in LDS
// Phase 3: ctx = P @ V^T-layout (WMMA, A-frags from LDS)
// LDS: 128KB f32 scores + 64KB bf16 P = 192KB (<= 320KB per WGP on CDNA5)
// ---------------------------------------------------------------------------
__global__ void __launch_bounds__(128)
attn_fused(const unsigned short* __restrict__ Qb,
           const unsigned short* __restrict__ Kb,
           const unsigned short* __restrict__ Vt,
           const int* __restrict__ mask,
           unsigned short* __restrict__ ctx) {
  const int q0   = blockIdx.x * 16;
  const int h    = blockIdx.y;
  const int b    = blockIdx.z;
  const int lane = threadIdx.x & 31;
  const int wv   = threadIdx.x >> 5;

  extern __shared__ char smem[];
  float*          sc = (float*)smem;                              // [16][SEQ]
  unsigned short* pb = (unsigned short*)(smem + 16 * SEQ * 4);    // [16][SEQ]

  const unsigned short* Qh = Qb + ((size_t)b * SEQ + q0) * DMODEL + h * DK;
  v16bf aQ0 = load_a_frag(Qh, DMODEL, 0, 0, lane);
  v16bf aQ1 = load_a_frag(Qh, DMODEL, 0, 32, lane);

  // Phase 1: scores
  const unsigned short* Kh = Kb + (size_t)b * SEQ * DMODEL + h * DK;
  for (int jt = wv; jt < SEQ / 16; jt += 4) {
    v8f acc = {};
    v16bf b0 = load_b_frag(Kh, DMODEL, jt * 16, 0, lane);
    acc = wmma_bf16(aQ0, b0, acc);
    v16bf b1 = load_b_frag(Kh, DMODEL, jt * 16, 32, lane);
    acc = wmma_bf16(aQ1, b1, acc);

    int scol = jt * 16 + (lane & 15);
    bool mz = (mask[(size_t)b * SEQ + scol] == 0);
    int rbase = (lane >> 4) * 8;
    #pragma unroll
    for (int r = 0; r < 8; r++) {
      float v = mz ? -1e20f : acc[r] * 0.125f;  // 1/sqrt(64)
      sc[(rbase + r) * SEQ + scol] = v;
    }
  }
  __syncthreads();

  // Phase 2: softmax (16 rows x 8 threads/row; row threads share a wave)
  {
    int row = threadIdx.x >> 3;
    int i   = threadIdx.x & 7;
    float* srow = sc + (size_t)row * SEQ;
    float mx = -3.4e38f;
    for (int c = i; c < SEQ; c += 8) mx = fmaxf(mx, srow[c]);
    #pragma unroll
    for (int off = 1; off < 8; off <<= 1) mx = fmaxf(mx, __shfl_xor(mx, off, 8));
    float sum = 0.0f;
    for (int c = i; c < SEQ; c += 8) {
      float e = __expf(srow[c] - mx);
      srow[c] = e;
      sum += e;
    }
    #pragma unroll
    for (int off = 1; off < 8; off <<= 1) sum += __shfl_xor(sum, off, 8);
    float inv = 1.0f / sum;
    unsigned short* prow = pb + (size_t)row * SEQ;
    for (int c = i; c < SEQ; c += 8) prow[c] = f2bf(srow[c] * inv);
  }
  __syncthreads();

  // Phase 3: ctx = P @ V  (V stored [B,H,Dk,S] -> contiguous B-frag loads)
  {
    v8f acc = {};
    const int d0 = wv * 16;
    const unsigned short* Vh = Vt + ((size_t)b * NHEADS + h) * DK * SEQ;
    for (int ks = 0; ks < SEQ; ks += 32) {
      v16bf a  = load_a_frag(pb, SEQ, 0, ks, lane);   // from LDS
      v16bf bb = load_b_frag(Vh, SEQ, d0, ks, lane);
      acc = wmma_bf16(a, bb, acc);
    }
    int nl = lane & 15;
    int rbase = (lane >> 4) * 8;
    #pragma unroll
    for (int r = 0; r < 8; r++) {
      int q = q0 + rbase + r;
      ctx[((size_t)b * SEQ + q) * DMODEL + h * DK + d0 + nl] = f2bf(acc[r]);
    }
  }
}

// ---------------------------------------------------------------------------
// LayerNorm over last dim (N=1024), one block per row.
// ---------------------------------------------------------------------------
__global__ void __launch_bounds__(256)
layernorm_k(const float* __restrict__ in, const float* __restrict__ g,
            const float* __restrict__ be, float* __restrict__ out_f,
            unsigned short* __restrict__ out_bf, int N) {
  const int row = blockIdx.x;
  const float* x = in + (size_t)row * N;
  __shared__ float red[256];
  int tid = threadIdx.x;

  float s = 0.0f;
  for (int c = tid; c < N; c += 256) s += x[c];
  red[tid] = s; __syncthreads();
  for (int off = 128; off > 0; off >>= 1) {
    if (tid < off) red[tid] += red[tid + off];
    __syncthreads();
  }
  float mu = red[0] / N;
  __syncthreads();

  float v = 0.0f;
  for (int c = tid; c < N; c += 256) { float d = x[c] - mu; v += d * d; }
  red[tid] = v; __syncthreads();
  for (int off = 128; off > 0; off >>= 1) {
    if (tid < off) red[tid] += red[tid + off];
    __syncthreads();
  }
  float rstd = rsqrtf(red[0] / N + 1e-5f);

  for (int c = tid; c < N; c += 256) {
    float y = (x[c] - mu) * rstd * g[c] + be[c];
    out_f[(size_t)row * N + c] = y;
    if (out_bf) out_bf[(size_t)row * N + c] = f2bf(y);
  }
}

// ---------------------------------------------------------------------------
// Host driver
// ---------------------------------------------------------------------------
extern "C" void kernel_launch(void* const* d_in, const int* in_sizes, int n_in,
                              void* d_out, int out_size, void* d_ws, size_t ws_size,
                              hipStream_t stream) {
  (void)in_sizes; (void)n_in; (void)out_size; (void)ws_size;
  const float* x   = (const float*)d_in[0];
  const int*  mask = (const int*)d_in[1];
  const float* Wq = (const float*)d_in[2];  const float* bq = (const float*)d_in[3];
  const float* Wk = (const float*)d_in[4];  const float* bk = (const float*)d_in[5];
  const float* Wv = (const float*)d_in[6];  const float* bv = (const float*)d_in[7];
  const float* Wo = (const float*)d_in[8];  const float* bo = (const float*)d_in[9];
  const float* W1 = (const float*)d_in[10]; const float* b1 = (const float*)d_in[11];
  const float* W2 = (const float*)d_in[12]; const float* b2 = (const float*)d_in[13];
  const float* g1 = (const float*)d_in[14]; const float* be1 = (const float*)d_in[15];
  const float* g2 = (const float*)d_in[16]; const float* be2 = (const float*)d_in[17];

  char* ws = (char*)d_ws;
  size_t off = 0;
  auto alloc = [&](size_t bytes) -> void* {
    void* p = ws + off;
    off = (off + bytes + 255) & ~(size_t)255;
    return p;
  };

  const size_t nX  = (size_t)MROWS * DMODEL;           // 8M
  const size_t nW  = (size_t)DMODEL * DMODEL;          // 1M
  const size_t nW1 = (size_t)DFF * DMODEL;             // 4M
  const size_t nFF = (size_t)MROWS * DFF;              // 32M

  unsigned short* xb  = (unsigned short*)alloc(nX * 2);
  unsigned short* Wqb = (unsigned short*)alloc(nW * 2);
  unsigned short* Wkb = (unsigned short*)alloc(nW * 2);
  unsigned short* Wvb = (unsigned short*)alloc(nW * 2);
  unsigned short* Wob = (unsigned short*)alloc(nW * 2);
  unsigned short* W1b = (unsigned short*)alloc(nW1 * 2);
  unsigned short* W2b = (unsigned short*)alloc(nW1 * 2);
  unsigned short* Qb  = (unsigned short*)alloc(nX * 2);
  unsigned short* Kb  = (unsigned short*)alloc(nX * 2);
  unsigned short* Vt  = (unsigned short*)alloc(nX * 2);   // [B,H,Dk,S]
  unsigned short* ctx = (unsigned short*)alloc(nX * 2);
  float*          res1 = (float*)alloc(nX * 4);
  float*          hbuf = (float*)alloc(nX * 4);
  unsigned short* hb   = (unsigned short*)alloc(nX * 2);
  unsigned short* ff1b = (unsigned short*)alloc(nFF * 2);
  float*          res2 = (float*)alloc(nX * 4);

  auto cvt = [&](const float* src, unsigned short* dst, size_t n) {
    int blocks = (int)((n + 256 * 4 - 1) / (256 * 4));
    cvt_f32_bf16<<<blocks, 256, 0, stream>>>(src, dst, n);
  };
  cvt(x, xb, nX);
  cvt(Wq, Wqb, nW); cvt(Wk, Wkb, nW); cvt(Wv, Wvb, nW); cvt(Wo, Wob, nW);
  cvt(W1, W1b, nW1); cvt(W2, W2b, nW1);

  dim3 blk(256);
  // --- QKV projections ----------------------------------------------------
  {
    dim3 grd(MROWS / 128, DMODEL / 128);
    gemm_bf16_wmma<0><<<grd, blk, 0, stream>>>(xb, Wqb, bq, Qb, nullptr,
                                               MROWS, DMODEL, DMODEL);
    gemm_bf16_wmma<0><<<grd, blk, 0, stream>>>(xb, Wkb, bk, Kb, nullptr,
                                               MROWS, DMODEL, DMODEL);
    gemm_bf16_wmma<1><<<grd, blk, 0, stream>>>(xb, Wvb, bv, Vt, nullptr,
                                               MROWS, DMODEL, DMODEL);
  }

  // --- Fused attention ----------------------------------------------------
  {
    size_t smem = (size_t)16 * SEQ * 4 + (size_t)16 * SEQ * 2;  // 192 KB
    hipFuncSetAttribute((const void*)attn_fused,
                        hipFuncAttributeMaxDynamicSharedMemorySize, (int)smem);
    dim3 grd(SEQ / 16, NHEADS, BATCH);
    attn_fused<<<grd, dim3(128), smem, stream>>>(Qb, Kb, Vt, mask, ctx);
  }

  // --- Output projection + residual --------------------------------------
  {
    dim3 grd(MROWS / 128, DMODEL / 128);
    gemm_bf16_wmma<2><<<grd, blk, 0, stream>>>(ctx, Wob, bo, res1, x,
                                               MROWS, DMODEL, DMODEL);
  }
  // --- LayerNorm 1 --------------------------------------------------------
  layernorm_k<<<MROWS, 256, 0, stream>>>(res1, g1, be1, hbuf, hb, DMODEL);

  // --- FFN ----------------------------------------------------------------
  {
    dim3 grd1(MROWS / 128, DFF / 128);
    gemm_bf16_wmma<3><<<grd1, blk, 0, stream>>>(hb, W1b, b1, ff1b, nullptr,
                                                MROWS, DFF, DMODEL);
    dim3 grd2(MROWS / 128, DMODEL / 128);
    gemm_bf16_wmma<2><<<grd2, blk, 0, stream>>>(ff1b, W2b, b2, res2, hbuf,
                                                MROWS, DMODEL, DFF);
  }
  // --- LayerNorm 2 -> output ---------------------------------------------
  layernorm_k<<<MROWS, 256, 0, stream>>>(res2, g2, be2, (float*)d_out,
                                         nullptr, DMODEL);
}
